// DTVIN_37821482009405
// MI455X (gfx1250) — matrix operand order repaired and checked by reading, not compiled, not run on hardware
//
#include <hip/hip_runtime.h>

// ---------------- problem constants ----------------
#define BATCH 128
#define CIN   2
#define HID   150
#define NA    8
#define NITER 30
#define GS    49            // GRID
#define HP    51            // GRID + 2
#define NCH   160           // HID padded to 5*32 (full 16x16x32 WMMA K-chunks)
#define NCHT  80            // 72 trans channels + 1 reward + pad -> 5*16 N-tiles
#define BHW   (GS*GS)       // 2401
#define BHWP  (HP*HP)       // 2601

typedef __attribute__((ext_vector_type(16))) _Float16 v16h;
typedef __attribute__((ext_vector_type(8)))  _Float16 v8h;
typedef __attribute__((ext_vector_type(8)))  float    v8f;

// ---------------- kernel 1: conv1 (C=2 -> 150) + relu, NCHW f32 -> NHWC f16 (padded 160) ----
__global__ void conv1_kernel(const float* __restrict__ g,    // [B][2][49][49]
                             const float* __restrict__ w,    // [150][2][3][3]
                             const float* __restrict__ bias, // [150]
                             _Float16* __restrict__ h1)      // [B][2401][160]
{
    int idx = blockIdx.x * blockDim.x + threadIdx.x;      // b*2401*160 + p*160 + co
    int co = idx % NCH;
    int bp = idx / NCH;
    int b = bp / BHW, p = bp % BHW;
    int y = p / GS, x = p % GS;
    float acc = 0.f;
    if (co < HID) {
        acc = bias[co];
        #pragma unroll
        for (int ci = 0; ci < CIN; ++ci) {
            #pragma unroll
            for (int t = 0; t < 9; ++t) {
                int iy = y + t / 3 - 1, ix = x + t % 3 - 1;
                if (iy >= 0 && iy < GS && ix >= 0 && ix < GS)
                    acc += g[((size_t)(b * CIN + ci) * GS + iy) * GS + ix] *
                           w[((co * CIN + ci) * 3 + t / 3) * 3 + t % 3];
            }
        }
        acc = fmaxf(acc, 0.f);
    }
    h1[(size_t)idx] = (_Float16)acc;
}

// ---------------- kernel 2: repack conv weights to f16 [tap][cout][cin]; zero the pad row ----
__global__ void repack_kernel(const float* __restrict__ h2w, // [150][150][3][3]
                              const float* __restrict__ tw,  // [72][150][3][3]
                              const float* __restrict__ rw,  // [1][150][3][3]
                              _Float16* __restrict__ w2p,    // [9][160][160]
                              _Float16* __restrict__ wtp,    // [9][80][160]
                              _Float16* __restrict__ zr)     // [160] zeros
{
    const int N2 = 9 * NCH * NCH;
    const int NT = 9 * NCHT * NCH;
    int idx = blockIdx.x * blockDim.x + threadIdx.x;
    if (idx < N2) {
        int i = idx % NCH, rest = idx / NCH;
        int o = rest % NCH, t = rest / NCH;
        _Float16 v = (_Float16)0.f;
        if (o < HID && i < HID)
            v = (_Float16)h2w[((o * HID + i) * 3 + t / 3) * 3 + t % 3];
        w2p[idx] = v;
    } else if (idx < N2 + NT) {
        int j = idx - N2;
        int i = j % NCH, rest = j / NCH;
        int o = rest % NCHT, t = rest / NCHT;
        _Float16 v = (_Float16)0.f;
        if (i < HID) {
            if (o < NA * 9)       v = (_Float16)tw[((o * HID + i) * 3 + t / 3) * 3 + t % 3];
            else if (o == NA * 9) v = (_Float16)rw[(i * 3 + t / 3) * 3 + t % 3]; // fused reward
        }
        wtp[j] = v;
    } else if (idx < N2 + NT + NCH) {
        zr[idx - N2 - NT] = (_Float16)0.f;     // zero row for out-of-bounds A-fragments
    }
}

// ---------------- implicit-GEMM 3x3 conv via v_wmma_f32_16x16x32_f16 ----------------
// src: NHWC f16 [B][49*49][160]; wp: f16 [9][NTO*16][160]
// Block = 8 waves sharing one ntile; per-block weight slab (45KB) staged into LDS
// with gfx1250 async global->LDS copies, drained by s_wait_asynccnt.
// One wave = one 16(pixels) x 16(outch) tile; K = 9 taps x 5 chunks of 32 = 45 WMMAs.
template<int OHW, int PAD, int NTO, bool BIAS_RELU>
__global__ void convK_wmma(const _Float16* __restrict__ src,
                           const _Float16* __restrict__ wp,
                           const float* __restrict__ bias,
                           const _Float16* __restrict__ zrow,
                           void* __restrict__ dst)
{
    constexpr int OPIX = OHW * OHW;
    constexpr int MT   = (OPIX + 15) / 16;
    constexpr int MG   = (MT + 7) / 8;       // mtile groups of 8 waves
    constexpr int NOUT = NTO * 16;
    constexpr int LROW = NCH + 8;            // 168 f16: padded LDS row vs. bank conflicts

    __shared__ _Float16 wlds[9 * 16 * LROW]; // ~47 KB of the 320 KB WGP LDS

    const int blk    = blockIdx.x;
    const int b      = blk / (MG * NTO);
    const int rem    = blk % (MG * NTO);
    const int mgroup = rem / NTO;
    const int ntile  = rem % NTO;

    // ---- async-stage this ntile's weights: [9][16 couts][160 cins] -> LDS ----
    {
        const int nchunks = 9 * 16 * (NCH / 8);          // 2880 x 16B
        for (int c = threadIdx.x; c < nchunks; c += 256) {
            int row = c / (NCH / 8);                     // tap*16 + cout_local
            int col = c % (NCH / 8);
            int t = row >> 4, cr = row & 15;
            const _Float16* gsrc =
                wp + ((size_t)t * NOUT + ntile * 16 + cr) * NCH + col * 8;
            unsigned lo = (unsigned)(unsigned long long)(const void*)
                              &wlds[row * LROW + col * 8];
            asm volatile("global_load_async_to_lds_b128 %0, %1, off"
                         :: "v"(lo), "v"(gsrc) : "memory");
        }
        asm volatile("s_wait_asynccnt 0x0" ::: "memory");
        __syncthreads();
    }

    const int wave  = threadIdx.x >> 5;
    const int lane  = threadIdx.x & 31;
    const int mtile = mgroup * 8 + wave;     // may exceed MT-1: neutralized via zrow + guards
    const int lrow  = lane & 15;             // A row (pixel) and B/D column (channel)
    const int khalf = lane >> 4;

    const int pix = mtile * 16 + lrow;
    const int py  = pix / OHW;
    const int px  = pix % OHW;
    const int cw  = ntile * 16 + lrow;       // output channel for B-frag / D store

    v8f acc = {};
    for (int t = 0; t < 9; ++t) {
        const int iy = py + t / 3 - PAD;
        const int ix = px + t % 3 - PAD;
        const bool pvalid = (pix < OPIX) && iy >= 0 && iy < GS && ix >= 0 && ix < GS;
        // branchless boundary handling: OOB lanes read a zeroed row (v_cndmask on address)
        const _Float16* arow = pvalid
            ? src + ((long)b * BHW + (long)iy * GS + ix) * NCH
            : zrow;
        const _Float16* brow = &wlds[(t * 16 + lrow) * LROW];
        #pragma unroll
        for (int c0 = 0; c0 < NCH; c0 += 32) {
            // A 16x32 f16 (ISA 7.12.2): lane half selects K 0-7/16-23 vs 8-15/24-31
            v8h lo = *(const v8h*)(arow + c0 + khalf * 8);
            v8h hi = *(const v8h*)(arow + c0 + 16 + khalf * 8);
            v16h a;
            #pragma unroll
            for (int e = 0; e < 8; ++e) { a[e] = lo[e]; a[8 + e] = hi[e]; }
            // B 32x16 f16 from LDS (ds_load_b128): lane half selects K 0-7/16-23 vs 8-15/24-31
            v8h blo = *(const v8h*)(brow + c0 + khalf * 8);
            v8h bhi = *(const v8h*)(brow + c0 + 16 + khalf * 8);
            v16h bf;
            #pragma unroll
            for (int e = 0; e < 8; ++e) { bf[e] = blo[e]; bf[8 + e] = bhi[e]; }
            acc = __builtin_amdgcn_wmma_f32_16x16x32_f16(
                false, a, false, bf, (short)0, acc, false, false);
        }
    }

    // D 16x16 f32: lane holds rows M = khalf*8 + r, column N = lane&15
    if (BIAS_RELU) {
        const float bs = (cw < HID) ? bias[cw] : 0.f;
        _Float16* dh = (_Float16*)dst;
        #pragma unroll
        for (int r = 0; r < 8; ++r) {
            int p = mtile * 16 + khalf * 8 + r;
            if (p < OPIX) {
                float v = acc[r] + bs;
                dh[((size_t)b * OPIX + p) * NOUT + cw] = (_Float16)(v > 0.f ? v : 0.f);
            }
        }
    } else {
        float* df = (float*)dst;
        #pragma unroll
        for (int r = 0; r < 8; ++r) {
            int p = mtile * 16 + khalf * 8 + r;
            if (p < OPIX)
                df[((size_t)b * OPIX + p) * NOUT + cw] = acc[r];
        }
    }
}

// ---------------- kernel 4: softmax over 9 taps; split out reward channel ----------------
__global__ void softmax_kernel(const float* __restrict__ tr,  // [B][2601][80]
                               _Float16* __restrict__ sT,     // [B][2601][72]
                               float* __restrict__ rew)       // [B][2601]
{
    int idx = blockIdx.x * blockDim.x + threadIdx.x;           // (b*2601 + p)*8 + a
    if (idx >= BATCH * BHWP * NA) return;
    int a = idx & 7;
    int bp = idx >> 3;
    const float* t = tr + (size_t)bp * NCHT + a * 9;
    float m = t[0];
    #pragma unroll
    for (int k = 1; k < 9; ++k) m = fmaxf(m, t[k]);
    float e[9], s = 0.f;
    #pragma unroll
    for (int k = 0; k < 9; ++k) { e[k] = __expf(t[k] - m); s += e[k]; }
    float inv = 1.f / s;
    #pragma unroll
    for (int k = 0; k < 9; ++k)
        sT[(size_t)bp * 72 + a * 9 + k] = (_Float16)(e[k] * inv);
    if (a == 0) rew[bp] = tr[(size_t)bp * NCHT + 72];          // fused reward channel
}

// ---------------- kernel 5: one value-iteration sweep ----------------
__global__ void vi_step(const _Float16* __restrict__ sT,  // [B][2601][72]
                        const float* __restrict__ rew,    // [B][2601]
                        const float* __restrict__ vin,    // [B][2601]
                        float* __restrict__ vout,         // [B][2601]
                        float* __restrict__ qout,         // [B][2601][8]
                        int write_q)
{
    int idx = blockIdx.x * blockDim.x + threadIdx.x;
    if (idx >= BATCH * BHWP) return;
    int b = idx / BHWP, p = idx % BHWP;
    int y = p / HP, x = p % HP;
    float patch[9];
    #pragma unroll
    for (int k = 0; k < 9; ++k) {
        int py = y + k / 3 - 1, px = x + k % 3 - 1;
        patch[k] = (py >= 0 && py < HP && px >= 0 && px < HP)
                       ? vin[(size_t)b * BHWP + py * HP + px] : 0.f;
    }
    float r = rew[idx];
    const _Float16* st = sT + (size_t)idx * 72;
    float vmax = -3.0e38f;
    #pragma unroll
    for (int a = 0; a < NA; ++a) {
        float q = r;
        #pragma unroll
        for (int k = 0; k < 9; ++k) q += (float)st[a * 9 + k] * patch[k];
        if (write_q) qout[(size_t)idx * NA + a] = q;
        vmax = fmaxf(vmax, q);
    }
    vout[idx] = vmax;
}

// ---------------- kernel 6: per-pixel 8 -> 150 -> 8 MLP head (crop 51->49) ----------------
__global__ void head_kernel(const float* __restrict__ qbuf, // [B][2601][8]
                            const float* __restrict__ a1w,  // [150][8]
                            const float* __restrict__ a1b,  // [150]
                            const float* __restrict__ a2w,  // [8][150]
                            const float* __restrict__ a2b,  // [8]
                            float* __restrict__ out)        // [B][8][49][49]
{
    int idx = blockIdx.x * blockDim.x + threadIdx.x;
    if (idx >= BATCH * BHW) return;
    int b = idx / BHW, p = idx % BHW;
    int y = p / GS, x = p % GS;
    const float* q = qbuf + ((size_t)b * BHWP + y * HP + x) * NA;
    float qr[NA], lg[NA];
    #pragma unroll
    for (int a = 0; a < NA; ++a) { qr[a] = q[a]; lg[a] = a2b[a]; }
    for (int c = 0; c < HID; ++c) {
        float m = a1b[c];
        #pragma unroll
        for (int a = 0; a < NA; ++a) m += qr[a] * a1w[c * NA + a];
        m = fmaxf(m, 0.f);
        #pragma unroll
        for (int a = 0; a < NA; ++a) lg[a] += m * a2w[a * HID + c];
    }
    #pragma unroll
    for (int a = 0; a < NA; ++a)
        out[((size_t)(b * NA + a) * GS + y) * GS + x] = lg[a];
}

__global__ void zero_f32(float* __restrict__ p, int n)
{
    int i = blockIdx.x * blockDim.x + threadIdx.x;
    if (i < n) p[i] = 0.f;
}

// ---------------- workspace layout ----------------
static constexpr size_t alnup(size_t x) { return (x + 255) & ~(size_t)255; }
constexpr size_t SZ_H1  = (size_t)BATCH * BHW * NCH * 2;        // 98.3 MB f16
constexpr size_t SZ_H2  = SZ_H1;
constexpr size_t SZ_W2P = (size_t)9 * NCH * NCH * 2;
constexpr size_t SZ_WTP = (size_t)9 * NCHT * NCH * 2;
constexpr size_t SZ_TR  = (size_t)BATCH * BHWP * NCHT * 4;      // 106.5 MB f32
constexpr size_t SZ_ST  = (size_t)BATCH * BHWP * 72 * 2;        // 47.9 MB f16
constexpr size_t SZ_RW  = (size_t)BATCH * BHWP * 4;
constexpr size_t SZ_V   = (size_t)BATCH * BHWP * 4;
constexpr size_t OFF_H1  = 0;
constexpr size_t OFF_H2  = alnup(OFF_H1 + SZ_H1);
constexpr size_t OFF_W2P = alnup(OFF_H2 + SZ_H2);
constexpr size_t OFF_WTP = alnup(OFF_W2P + SZ_W2P);
constexpr size_t OFF_ZR  = alnup(OFF_WTP + SZ_WTP);
constexpr size_t OFF_TR  = alnup(OFF_ZR + (size_t)NCH * 2);
constexpr size_t OFF_ST  = alnup(OFF_TR + SZ_TR);
constexpr size_t OFF_RW  = alnup(OFF_ST + SZ_ST);
constexpr size_t OFF_V0  = alnup(OFF_RW + SZ_RW);
constexpr size_t OFF_V1  = alnup(OFF_V0 + SZ_V);
constexpr size_t OFF_Q   = alnup(OFF_V1 + SZ_V);

extern "C" void kernel_launch(void* const* d_in, const int* in_sizes, int n_in,
                              void* d_out, int out_size, void* d_ws, size_t ws_size,
                              hipStream_t stream)
{
    const float* gridp = (const float*)d_in[0];
    // d_in[1]=x_coord, d_in[2]=y_coord are unused by the reference
    const float* h1w = (const float*)d_in[3];
    const float* h1b = (const float*)d_in[4];
    const float* h2w = (const float*)d_in[5];
    const float* h2b = (const float*)d_in[6];
    const float* rw  = (const float*)d_in[7];
    const float* tw  = (const float*)d_in[8];
    const float* a1w = (const float*)d_in[9];
    const float* a1b = (const float*)d_in[10];
    const float* a2w = (const float*)d_in[11];
    const float* a2b = (const float*)d_in[12];

    char* ws = (char*)d_ws;
    _Float16* h1  = (_Float16*)(ws + OFF_H1);
    _Float16* h2  = (_Float16*)(ws + OFF_H2);
    _Float16* w2p = (_Float16*)(ws + OFF_W2P);
    _Float16* wtp = (_Float16*)(ws + OFF_WTP);
    _Float16* zr  = (_Float16*)(ws + OFF_ZR);
    float*    trb = (float*)(ws + OFF_TR);
    _Float16* sT  = (_Float16*)(ws + OFF_ST);
    float*    rew = (float*)(ws + OFF_RW);
    float*    v0  = (float*)(ws + OFF_V0);
    float*    v1  = (float*)(ws + OFF_V1);
    float*    qb  = (float*)(ws + OFF_Q);

    // conv1 + relu -> h1 (NHWC f16, padded channels)
    {
        int n = BATCH * BHW * NCH;                       // 49,172,480
        conv1_kernel<<<n / 256, 256, 0, stream>>>(gridp, h1w, h1b, h1);
    }
    // repack f16 weights for implicit GEMM + zero row
    {
        int n = 9 * NCH * NCH + 9 * NCHT * NCH + NCH;
        repack_kernel<<<(n + 255) / 256, 256, 0, stream>>>(h2w, tw, rw, w2p, wtp, zr);
    }
    // conv2 + bias + relu -> h2 (WMMA implicit GEMM, 124.5 GFLOP)
    {
        constexpr int MT = (BHW + 15) / 16;              // 151
        constexpr int MG = (MT + 7) / 8;                 // 19
        int blocks = BATCH * MG * (NCH / 16);            // 24,320
        convK_wmma<GS, 1, NCH / 16, true>
            <<<blocks, 256, 0, stream>>>(h1, w2p, h2b, zr, h2);
    }
    // trans conv (72 ch) + fused reward (ch 72), pad=2 -> 51x51 (WMMA, 65.5 GFLOP)
    {
        constexpr int MT = (BHWP + 15) / 16;             // 163
        constexpr int MG = (MT + 7) / 8;                 // 21
        int blocks = BATCH * MG * (NCHT / 16);           // 13,440
        convK_wmma<HP, 2, NCHT / 16, false>
            <<<blocks, 256, 0, stream>>>(h2, wtp, nullptr, zr, trb);
    }
    // softmax over taps + reward split
    {
        int n = BATCH * BHWP * NA;                       // 2,663,424
        softmax_kernel<<<(n + 255) / 256, 256, 0, stream>>>(trb, sT, rew);
    }
    // value iteration: v starts at zero, 30 sweeps, keep q from last sweep
    {
        int n = BATCH * BHWP;                            // 332,928
        int blks = (n + 255) / 256;
        zero_f32<<<blks, 256, 0, stream>>>(v0, n);
        float* va = v0;
        float* vb = v1;
        for (int i = 0; i < NITER; ++i) {
            vi_step<<<blks, 256, 0, stream>>>(sT, rew, va, vb, qb, (i == NITER - 1) ? 1 : 0);
            float* t = va; va = vb; vb = t;
        }
    }
    // per-pixel MLP head -> logits (B,8,49,49) f32
    {
        int n = BATCH * BHW;                             // 307,328
        head_kernel<<<(n + 255) / 256, 256, 0, stream>>>(qb, a1w, a1b, a2w, a2b, (float*)d_out);
    }
    (void)in_sizes; (void)n_in; (void)out_size; (void)ws_size;
}
